// CausalSelfAttention_40999757808142
// MI455X (gfx1250) — compile-verified, matrix operand order checked by reading, not run on hardware
//
#include <hip/hip_runtime.h>

typedef __attribute__((ext_vector_type(16))) __bf16 v16bf;
typedef __attribute__((ext_vector_type(8)))  __bf16 v8bf;
typedef __attribute__((ext_vector_type(4)))  __bf16 v4bf;
typedef __attribute__((ext_vector_type(8)))  float  v8f;

#define BDIM 4
#define TDIM 2048
#define CDIM 1024
#define NHDIM 16
#define HDDIM 64

// ---------- helpers ----------

__device__ __forceinline__ __bf16 f2bf(float x) {
  union { float f; unsigned u; } a; a.f = x;
  unsigned r = a.u + 0x7FFFu + ((a.u >> 16) & 1u);   // round-to-nearest-even
  union { unsigned short s; __bf16 b; } c;
  c.s = (unsigned short)(r >> 16);
  return c.b;
}

// A/B fragment loader for 16x16x32 bf16 WMMA from a row-major matrix.
// lane L: row/col = L&15 ; chunk0 = K[klo..klo+7], chunk1 = K[klo+16..klo+23],
// klo = (L<16) ? 0 : 8.  Two 16-byte loads per lane.
__device__ __forceinline__ v16bf load_frag(const __bf16* p, int ld) {
  const int lane = threadIdx.x & 31;
  const int row  = lane & 15;
  const int klo  = (lane < 16) ? 0 : 8;
  const __bf16* q = p + (size_t)row * ld + klo;
  union { v16bf v; struct { v8bf lo; v8bf hi; } s; } u;
  u.s.lo = *(const v8bf*)(q);
  u.s.hi = *(const v8bf*)(q + 16);
  return u.v;
}

__device__ __forceinline__ v8f wmma_bf16(v16bf a, v16bf b, v8f c) {
  return __builtin_amdgcn_wmma_f32_16x16x32_bf16(false, a, false, b, (short)0, c,
                                                 false, false);
}

__device__ __forceinline__ float rowmax16(float v) {
  v = fmaxf(v, __shfl_xor(v, 1, 32));
  v = fmaxf(v, __shfl_xor(v, 2, 32));
  v = fmaxf(v, __shfl_xor(v, 4, 32));
  v = fmaxf(v, __shfl_xor(v, 8, 32));
  return v;
}
__device__ __forceinline__ float rowsum16(float v) {
  v += __shfl_xor(v, 1, 32);
  v += __shfl_xor(v, 2, 32);
  v += __shfl_xor(v, 4, 32);
  v += __shfl_xor(v, 8, 32);
  return v;
}

// ---------- stage 0: fp32 -> bf16 convert ----------

__global__ __launch_bounds__(256) void cvt_f32_bf16(const float* __restrict__ in,
                                                    __bf16* __restrict__ out, int n4) {
  int i = blockIdx.x * blockDim.x + threadIdx.x;
  if (i >= n4) return;
  float4 v = ((const float4*)in)[i];
  v4bf o;
  o[0] = f2bf(v.x); o[1] = f2bf(v.y); o[2] = f2bf(v.z); o[3] = f2bf(v.w);
  ((v4bf*)out)[i] = o;
}

// ---------- stage 1: QKV projection GEMM ----------
// out[m][n] = sum_k x[m][k] * W[n][k] + bias[n]  (M=8192, N=3072, K=1024)
// scattered to q/k: [B,NH,T,HD], v transposed: [B,NH,HD,T], all bf16.

__global__ __launch_bounds__(256) void qkv_gemm(const __bf16* __restrict__ xb,
                                                const __bf16* __restrict__ wb,
                                                const float* __restrict__ bias,
                                                __bf16* __restrict__ qo,
                                                __bf16* __restrict__ ko,
                                                __bf16* __restrict__ vto) {
  const int lane = threadIdx.x & 31;
  const int wave = threadIdx.x >> 5;
  const int wm = wave >> 1, wn = wave & 1;
  const int m0 = blockIdx.y * 128 + wm * 32;
  const int n0 = blockIdx.x * 128 + wn * 64;
  const int K  = CDIM;

  v8f zero = {};
  v8f acc[2][4];
#pragma unroll
  for (int i = 0; i < 2; ++i)
#pragma unroll
    for (int j = 0; j < 4; ++j) acc[i][j] = zero;

  for (int k0 = 0; k0 < K; k0 += 32) {
    v16bf a0 = load_frag(xb + (size_t)m0 * K + k0, K);
    v16bf a1 = load_frag(xb + (size_t)(m0 + 16) * K + k0, K);
#pragma unroll
    for (int nt = 0; nt < 4; ++nt) {
      v16bf bf = load_frag(wb + (size_t)(n0 + nt * 16) * K + k0, K);
      acc[0][nt] = wmma_bf16(a0, bf, acc[0][nt]);
      acc[1][nt] = wmma_bf16(a1, bf, acc[1][nt]);
    }
  }

  const int hi = (lane >= 16) ? 8 : 0;
  const int nn = lane & 15;
#pragma unroll
  for (int nt = 0; nt < 4; ++nt) {
    const int col   = n0 + nt * 16 + nn;
    const float bv  = bias[col];
    const int which = col >> 10;          // 0=q 1=k 2=v
    const int cc    = col & (CDIM - 1);
    const int h     = cc >> 6, d = cc & (HDDIM - 1);
#pragma unroll
    for (int mi = 0; mi < 2; ++mi) {
#pragma unroll
      for (int r = 0; r < 8; ++r) {
        const int rowg = m0 + mi * 16 + r + hi;
        const int bb = rowg >> 11, t = rowg & (TDIM - 1);
        const __bf16 val = f2bf(acc[mi][nt][r] + bv);
        if (which == 0)
          qo[(((size_t)(bb * NHDIM + h)) * TDIM + t) * HDDIM + d] = val;
        else if (which == 1)
          ko[(((size_t)(bb * NHDIM + h)) * TDIM + t) * HDDIM + d] = val;
        else
          vto[(((size_t)(bb * NHDIM + h)) * HDDIM + d) * TDIM + t] = val;
      }
    }
  }
}

// ---------- stage 2: causal flash attention ----------
// One wave owns 16 query rows. 8 waves/block -> 128 rows. grid = B*NH*(T/128).

#define LDP 40  // LDS row stride in bf16 (16B aligned, bank-staggered)

__global__ __launch_bounds__(256) void attn_fwd(const __bf16* __restrict__ q,
                                                const __bf16* __restrict__ k,
                                                const __bf16* __restrict__ vt,
                                                __bf16* __restrict__ y) {
  __shared__ __bf16 pls[8][16 * LDP];

  const int lane = threadIdx.x & 31;
  const int wave = threadIdx.x >> 5;
  const int nblkT = TDIM / 128;
  const int bh   = blockIdx.x / nblkT;
  const int qblk = blockIdx.x % nblkT;
  const int b = bh / NHDIM, h = bh % NHDIM;
  const int qb = qblk * 128 + wave * 16;

  const __bf16* Q  = q  + ((size_t)(b * NHDIM + h)) * TDIM * HDDIM;
  const __bf16* Kp = k  + ((size_t)(b * NHDIM + h)) * TDIM * HDDIM;
  const __bf16* Vt = vt + ((size_t)(b * NHDIM + h)) * HDDIM * TDIM;
  __bf16* lw = &pls[wave][0];

  const v16bf aq0 = load_frag(Q + (size_t)qb * HDDIM, HDDIM);
  const v16bf aq1 = load_frag(Q + (size_t)qb * HDDIM + 32, HDDIM);

  const int hi = (lane >= 16) ? 8 : 0;
  const int nn = lane & 15;

  float m_i[8], l_i[8];
#pragma unroll
  for (int r = 0; r < 8; ++r) { m_i[r] = -1e30f; l_i[r] = 0.0f; }
  v8f o0 = {}, o1 = {}, o2 = {}, o3 = {};

  const float scale = 0.125f;  // 1/sqrt(64)

  for (int j0 = 0; j0 < qb + 16; j0 += 32) {
    // S = Q K^T for 32 key columns (2 tiles x 2 k-steps)
    v8f s0 = {}, s1 = {};
    {
      v16bf bk;
      bk = load_frag(Kp + (size_t)j0 * HDDIM, HDDIM);        s0 = wmma_bf16(aq0, bk, s0);
      bk = load_frag(Kp + (size_t)j0 * HDDIM + 32, HDDIM);   s0 = wmma_bf16(aq1, bk, s0);
      bk = load_frag(Kp + (size_t)(j0 + 16) * HDDIM, HDDIM);      s1 = wmma_bf16(aq0, bk, s1);
      bk = load_frag(Kp + (size_t)(j0 + 16) * HDDIM + 32, HDDIM); s1 = wmma_bf16(aq1, bk, s1);
    }

    float p0[8], p1[8], alpha[8];
#pragma unroll
    for (int r = 0; r < 8; ++r) {
      const int rowg = qb + r + hi;
      float v0 = s0[r] * scale;
      float v1 = s1[r] * scale;
      if (j0 + nn > rowg)      v0 = -1e30f;   // causal mask
      if (j0 + 16 + nn > rowg) v1 = -1e30f;
      const float mx   = rowmax16(fmaxf(v0, v1));
      const float mnew = fmaxf(m_i[r], mx);
      alpha[r] = __expf(m_i[r] - mnew);
      p0[r]    = __expf(v0 - mnew);
      p1[r]    = __expf(v1 - mnew);
      l_i[r]   = l_i[r] * alpha[r] + rowsum16(p0[r] + p1[r]);
      m_i[r]   = mnew;
    }

    // rescale O, stage P (C/D layout -> A layout) through per-wave LDS tile
#pragma unroll
    for (int r = 0; r < 8; ++r) {
      o0[r] *= alpha[r]; o1[r] *= alpha[r]; o2[r] *= alpha[r]; o3[r] *= alpha[r];
      lw[(r + hi) * LDP + nn]      = f2bf(p0[r]);
      lw[(r + hi) * LDP + 16 + nn] = f2bf(p1[r]);
    }
    const v16bf ap = load_frag(lw, LDP);   // same-wave LDS ops are in-order

    // O += P V  (V transposed: Vt[d][t])
    o0 = wmma_bf16(ap, load_frag(Vt + (size_t)(0 * 16) * TDIM + j0, TDIM), o0);
    o1 = wmma_bf16(ap, load_frag(Vt + (size_t)(1 * 16) * TDIM + j0, TDIM), o1);
    o2 = wmma_bf16(ap, load_frag(Vt + (size_t)(2 * 16) * TDIM + j0, TDIM), o2);
    o3 = wmma_bf16(ap, load_frag(Vt + (size_t)(3 * 16) * TDIM + j0, TDIM), o3);
  }

#pragma unroll
  for (int r = 0; r < 8; ++r) {
    const float inv = 1.0f / l_i[r];
    const int t = qb + r + hi;
    __bf16* dst = y + (((size_t)b * TDIM + t)) * CDIM + h * HDDIM + nn;
    dst[0]  = f2bf(o0[r] * inv);
    dst[16] = f2bf(o1[r] * inv);
    dst[32] = f2bf(o2[r] * inv);
    dst[48] = f2bf(o3[r] * inv);
  }
}

// ---------- stage 3: output projection GEMM ----------
// out[m][n] = sum_k y[m][k] * Wp[n][k] + bias[n], fp32 output (M=8192, N=K=1024)

__global__ __launch_bounds__(256) void proj_gemm(const __bf16* __restrict__ yb,
                                                 const __bf16* __restrict__ wb,
                                                 const float* __restrict__ bias,
                                                 float* __restrict__ out) {
  const int lane = threadIdx.x & 31;
  const int wave = threadIdx.x >> 5;
  const int wm = wave >> 1, wn = wave & 1;
  const int m0 = blockIdx.y * 128 + wm * 32;
  const int n0 = blockIdx.x * 128 + wn * 64;
  const int K  = CDIM;

  v8f zero = {};
  v8f acc[2][4];
#pragma unroll
  for (int i = 0; i < 2; ++i)
#pragma unroll
    for (int j = 0; j < 4; ++j) acc[i][j] = zero;

  for (int k0 = 0; k0 < K; k0 += 32) {
    v16bf a0 = load_frag(yb + (size_t)m0 * K + k0, K);
    v16bf a1 = load_frag(yb + (size_t)(m0 + 16) * K + k0, K);
#pragma unroll
    for (int nt = 0; nt < 4; ++nt) {
      v16bf bf = load_frag(wb + (size_t)(n0 + nt * 16) * K + k0, K);
      acc[0][nt] = wmma_bf16(a0, bf, acc[0][nt]);
      acc[1][nt] = wmma_bf16(a1, bf, acc[1][nt]);
    }
  }

  const int hi = (lane >= 16) ? 8 : 0;
  const int nn = lane & 15;
#pragma unroll
  for (int nt = 0; nt < 4; ++nt) {
    const int col  = n0 + nt * 16 + nn;
    const float bv = bias[col];
#pragma unroll
    for (int mi = 0; mi < 2; ++mi) {
#pragma unroll
      for (int r = 0; r < 8; ++r) {
        const int rowg = m0 + mi * 16 + r + hi;
        out[(size_t)rowg * CDIM + col] = acc[mi][nt][r] + bv;
      }
    }
  }
}

// ---------- launch ----------

extern "C" void kernel_launch(void* const* d_in, const int* in_sizes, int n_in,
                              void* d_out, int out_size, void* d_ws, size_t ws_size,
                              hipStream_t stream) {
  const float* x      = (const float*)d_in[0];
  const float* W_attn = (const float*)d_in[1];
  const float* b_attn = (const float*)d_in[2];
  const float* W_proj = (const float*)d_in[3];
  const float* b_proj = (const float*)d_in[4];
  float* out = (float*)d_out;

  const size_t nBTC = (size_t)BDIM * TDIM * CDIM;      // 8,388,608
  const size_t nWA  = (size_t)3 * CDIM * CDIM;          // 3,145,728
  const size_t nWP  = (size_t)CDIM * CDIM;              // 1,048,576

  char* ws = (char*)d_ws;
  size_t off = 0;
  __bf16* xb   = (__bf16*)(ws + off); off += nBTC * 2;
  __bf16* wab  = (__bf16*)(ws + off); off += nWA * 2;
  __bf16* wpb  = (__bf16*)(ws + off); off += nWP * 2;
  __bf16* qws  = (__bf16*)(ws + off); off += nBTC * 2;
  __bf16* kws  = (__bf16*)(ws + off); off += nBTC * 2;
  __bf16* vtws = (__bf16*)(ws + off); off += nBTC * 2;
  __bf16* yws  = (__bf16*)(ws + off); off += nBTC * 2;
  (void)off; (void)ws_size; (void)in_sizes; (void)n_in; (void)out_size;

  // stage 0: convert fp32 -> bf16
  cvt_f32_bf16<<<(int)(nBTC / 4 + 255) / 256, 256, 0, stream>>>(x, xb, (int)(nBTC / 4));
  cvt_f32_bf16<<<(int)(nWA / 4 + 255) / 256, 256, 0, stream>>>(W_attn, wab, (int)(nWA / 4));
  cvt_f32_bf16<<<(int)(nWP / 4 + 255) / 256, 256, 0, stream>>>(W_proj, wpb, (int)(nWP / 4));

  // stage 1: QKV projection
  qkv_gemm<<<dim3((3 * CDIM) / 128, (BDIM * TDIM) / 128), 256, 0, stream>>>(
      xb, wab, b_attn, qws, kws, vtws);

  // stage 2: causal flash attention
  attn_fwd<<<dim3(BDIM * NHDIM * (TDIM / 128)), 256, 0, stream>>>(qws, kws, vtws, yws);

  // stage 3: output projection
  proj_gemm<<<dim3(CDIM / 128, (BDIM * TDIM) / 128), 256, 0, stream>>>(
      yws, wpb, b_proj, out);
}
